// ODENet_54984171323394
// MI455X (gfx1250) — compile-verified, matrix operand order checked by reading, not run on hardware
//
#include <hip/hip_runtime.h>
#include <hip/hip_bf16.h>

typedef _Float16 v8h  __attribute__((ext_vector_type(8)));
typedef _Float16 v16h __attribute__((ext_vector_type(16)));
typedef float    v8f  __attribute__((ext_vector_type(8)));
typedef int      i4v  __attribute__((ext_vector_type(4)));

#define BATCH 256
#define NNODES 512
#define FEAT 64
#define NELEM ((size_t)BATCH * NNODES * FEAT)   // 8388608

// ---------------------------------------------------------------------------
// Async global->LDS copy (CDNA5 GLOBAL_LOAD_ASYNC_TO_LDS_B128, ASYNCcnt) with
// a synchronous fallback if the builtin is unavailable on this toolchain.
// Probe result: builtin signature is (int4 AS(1)*, int4 AS(3)*, imm, imm).
// ---------------------------------------------------------------------------
#if defined(__gfx1250__) && __has_builtin(__builtin_amdgcn_global_load_async_to_lds_b128)
#define ASYNC_LDS 1
typedef __attribute__((address_space(1))) i4v* as1_i4p;
typedef __attribute__((address_space(3))) i4v* as3_i4p;
__device__ __forceinline__ void async_copy16(const _Float16* g, _Float16* l) {
    __builtin_amdgcn_global_load_async_to_lds_b128(
        (as1_i4p)(unsigned long long)g,
        (as3_i4p)(unsigned)(unsigned long long)l, 0, 0);
}
__device__ __forceinline__ void async_wait() {
    asm volatile("s_wait_asynccnt 0x0" ::: "memory");
}
#else
__device__ __forceinline__ void async_copy16(const _Float16* g, _Float16* l) {
    *(v8h*)l = *(const v8h*)g;
}
__device__ __forceinline__ void async_wait() {}
#endif

// Load a 16-half WMMA A/B fragment from an LDS row with k-contiguous halves.
// lanes 0-15: K = {koff..koff+7, koff+16..koff+23} with koff=0;
// lanes 16-31: koff=8.  Two ds_load_b128 per lane.
__device__ __forceinline__ v16h load_frag(const _Float16* rowp, int koff) {
    v8h lo = *(const v8h*)(rowp + koff);
    v8h hi = *(const v8h*)(rowp + koff + 16);
    return __builtin_shufflevector(lo, hi, 0,1,2,3,4,5,6,7,8,9,10,11,12,13,14,15);
}

__device__ __forceinline__ v8f wmma_f16(v16h a, v16h b, v8f c) {
    return __builtin_amdgcn_wmma_f32_16x16x32_f16(false, a, false, b, (short)0, c,
                                                  false, false);
}

// Fill one double-buffered k-chunk: A tile (256 rows x 32 k, async copies) and
// transposed X tile Xt[f][k] (ds_store transpose).
__device__ __forceinline__ void fill_tile(
    const _Float16* __restrict__ Ah, const _Float16* __restrict__ Xin_b,
    int row0, int k0, int tid, _Float16 (*As)[40], _Float16 (*Xt)[40])
{
    const int row = tid >> 2, seg = (tid & 3) * 8;   // 4 lanes per 64B row chunk
    #pragma unroll
    for (int rr = 0; rr < 4; ++rr)
        async_copy16(Ah + (size_t)(row0 + rr * 64 + row) * NNODES + k0 + seg,
                     &As[rr * 64 + row][seg]);
    const int xk = tid >> 3, xf0 = (tid & 7) * 8;
    v8h xv = *(const v8h*)(Xin_b + (size_t)(k0 + xk) * FEAT + xf0);
    #pragma unroll
    for (int j = 0; j < 8; ++j) Xt[xf0 + j][xk] = xv[j];
}

// ---------------------------------------------------------------------------
// out[b] = relu( (A @ X[b]) @ W )  for one 256-row strip of one batch.
// 8 waves; each wave owns a 32x64 output strip (2 A-frags reuse 4 B-frags).
// Double-buffered LDS tiles, async fills overlapped with WMMA compute.
// ---------------------------------------------------------------------------
__global__ __launch_bounds__(256) void layer_kernel(
    const _Float16* __restrict__ Xin, const _Float16* __restrict__ Ah,
    const _Float16* __restrict__ Wl,
    float* __restrict__ outF32, _Float16* __restrict__ outF16)
{
    __shared__ _Float16 As[2][256][40];   // 80B row stride (16B aligned, no conflicts)
    __shared__ _Float16 Xt[2][64][40];    // Xt[f][k]
    __shared__ _Float16 Wt[64][72];       // Wt[g][k] (144B stride)
    __shared__ _Float16 Tt[256][72];      // (A@X) staged f16, k-contiguous

    const int strip = blockIdx.x;         // 0..1
    const int b     = blockIdx.y;         // 0..255
    const int tid   = threadIdx.x;
    const int lane  = tid & 31;
    const int w     = tid >> 5;
    const int row0  = strip * 256;
    const int wr    = w * 32;             // wave's row strip inside tile
    const _Float16* Xin_b = Xin + (size_t)b * NNODES * FEAT;

    // stage W transposed: Wt[g][k] = W[k][g]
    {
        int g = tid & 63, kb = (tid >> 6) * 16;
        for (int j = 0; j < 16; ++j) Wt[g][kb + j] = Wl[(kb + j) * 64 + g];
    }

    v8f acc0[4] = {v8f{}, v8f{}, v8f{}, v8f{}};
    v8f acc1[4] = {v8f{}, v8f{}, v8f{}, v8f{}};

    const int lr   = lane & 15;
    const int koff = (lane < 16) ? 0 : 8;
    const int mhi  = (lane < 16) ? 0 : 8;

    // prologue: tile 0 into buffer 0
    fill_tile(Ah, Xin_b, row0, 0, tid, As[0], Xt[0]);
    async_wait();
    __syncthreads();

    int bb = 0;
    for (int kt = 0; kt < 16; ++kt) {
        if (kt < 15)   // prefetch next k-chunk into the other buffer
            fill_tile(Ah, Xin_b, row0, (kt + 1) * 32, tid, As[bb ^ 1], Xt[bb ^ 1]);

        v16h a0 = load_frag(&As[bb][wr + lr][0], koff);
        v16h a1 = load_frag(&As[bb][wr + 16 + lr][0], koff);
        #pragma unroll
        for (int t = 0; t < 4; ++t) {
            v16h bf = load_frag(&Xt[bb][t * 16 + lr][0], koff);
            acc0[t] = wmma_f16(a0, bf, acc0[t]);
            acc1[t] = wmma_f16(a1, bf, acc1[t]);
        }
        async_wait();
        __syncthreads();
        bb ^= 1;
    }

    // stage T = A@X (f16); each wave writes/reads only its own 32 rows
    #pragma unroll
    for (int t = 0; t < 4; ++t)
        #pragma unroll
        for (int j = 0; j < 8; ++j) {
            Tt[wr + mhi + j][t * 16 + lr]      = (_Float16)acc0[t][j];
            Tt[wr + 16 + mhi + j][t * 16 + lr] = (_Float16)acc1[t][j];
        }
    __syncthreads();

    // second GEMM: D = T @ W  (K = 64 -> two k-chunks)
    v8f d0[4] = {v8f{}, v8f{}, v8f{}, v8f{}};
    v8f d1[4] = {v8f{}, v8f{}, v8f{}, v8f{}};
    #pragma unroll
    for (int kc = 0; kc < 64; kc += 32) {
        v16h a20 = load_frag(&Tt[wr + lr][kc], koff);
        v16h a21 = load_frag(&Tt[wr + 16 + lr][kc], koff);
        #pragma unroll
        for (int t = 0; t < 4; ++t) {
            v16h bf = load_frag(&Wt[t * 16 + lr][kc], koff);
            d0[t] = wmma_f16(a20, bf, d0[t]);
            d1[t] = wmma_f16(a21, bf, d1[t]);
        }
    }

    // relu + store
    const int gm = row0 + wr + mhi;
    #pragma unroll
    for (int t = 0; t < 4; ++t) {
        int col = t * 16 + lr;
        #pragma unroll
        for (int j = 0; j < 8; ++j) {
            float v0 = d0[t][j]; v0 = v0 > 0.f ? v0 : 0.f;
            float v1 = d1[t][j]; v1 = v1 > 0.f ? v1 : 0.f;
            size_t i0 = ((size_t)b * NNODES + gm + j) * FEAT + col;
            size_t i1 = ((size_t)b * NNODES + gm + 16 + j) * FEAT + col;
            if (outF32) { outF32[i0] = v0; outF32[i1] = v1; }
            if (outF16) { outF16[i0] = (_Float16)v0; outF16[i1] = (_Float16)v1; }
        }
    }
}

// ---------------------------------------------------------------------------
// RK4 bookkeeping. mode 0: acc = x + ca*h*K ; Xh = f16(x + cb*h*K)
//                  mode 1: acc += ca*h*K    ; Xh = f16(x + cb*h*K)
//                  mode 2: x = acc + ca*h*K ; Xh = f16(x)
// ---------------------------------------------------------------------------
__global__ __launch_bounds__(256) void update_kernel(
    float* __restrict__ x, float* __restrict__ acc, const float* __restrict__ K,
    _Float16* __restrict__ Xh, const float* __restrict__ t0p,
    const float* __restrict__ t1p, float ca, float cb, int mode)
{
    size_t i = ((size_t)blockIdx.x * 256 + threadIdx.x) * 4;
    float h = (t1p[0] - t0p[0]) * 0.1f;
    float4 kv = *(const float4*)(K + i);
    if (mode == 2) {
        float4 av = *(const float4*)(acc + i);
        float c = ca * h;
        float4 xn = {av.x + c * kv.x, av.y + c * kv.y, av.z + c * kv.z, av.w + c * kv.w};
        *(float4*)(x + i) = xn;
        Xh[i]     = (_Float16)xn.x; Xh[i + 1] = (_Float16)xn.y;
        Xh[i + 2] = (_Float16)xn.z; Xh[i + 3] = (_Float16)xn.w;
    } else {
        float4 xv = *(const float4*)(x + i);
        float c1 = ca * h, c2 = cb * h;
        float4 av;
        if (mode == 0) {
            av = {xv.x + c1 * kv.x, xv.y + c1 * kv.y, xv.z + c1 * kv.z, xv.w + c1 * kv.w};
        } else {
            av = *(const float4*)(acc + i);
            av.x += c1 * kv.x; av.y += c1 * kv.y; av.z += c1 * kv.z; av.w += c1 * kv.w;
        }
        *(float4*)(acc + i) = av;
        Xh[i]     = (_Float16)(xv.x + c2 * kv.x);
        Xh[i + 1] = (_Float16)(xv.y + c2 * kv.y);
        Xh[i + 2] = (_Float16)(xv.z + c2 * kv.z);
        Xh[i + 3] = (_Float16)(xv.w + c2 * kv.w);
    }
}

// ---------------------------------------------------------------------------
// Row normalization of Z=[H0,H] (ddof=1) -> Zn f16 and Pw = Zn * w f16.
// ---------------------------------------------------------------------------
__global__ __launch_bounds__(256) void norm_kernel(
    const float* __restrict__ h0, const float* __restrict__ x,
    const float* __restrict__ wvec, _Float16* __restrict__ Zn,
    _Float16* __restrict__ Pw)
{
    int row  = blockIdx.x * 8 + (threadIdx.x >> 5);
    int lane = threadIdx.x & 31;
    const float* s0 = h0 + (size_t)row * FEAT;
    const float* s1 = x  + (size_t)row * FEAT;
    float v[4], s = 0.f, ss = 0.f;
    #pragma unroll
    for (int i = 0; i < 4; ++i) {
        int d = lane + 32 * i;
        float val = (d < 64) ? s0[d] : s1[d - 64];
        v[i] = val; s += val; ss += val * val;
    }
    #pragma unroll
    for (int m = 16; m >= 1; m >>= 1) {
        s  += __shfl_xor(s,  m, 32);
        ss += __shfl_xor(ss, m, 32);
    }
    float mu  = s * (1.0f / 128.0f);
    float var = (ss - 128.0f * mu * mu) * (1.0f / 127.0f);
    float inv = rsqrtf(var > 0.f ? var : 1e-30f);
    #pragma unroll
    for (int i = 0; i < 4; ++i) {
        int d = lane + 32 * i;
        float z = (v[i] - mu) * inv;
        Zn[(size_t)row * 128 + d] = (_Float16)z;
        Pw[(size_t)row * 128 + d] = (_Float16)(z * wvec[d]);
    }
}

// ---------------------------------------------------------------------------
// S[b] = Pw[b] @ Zn[b]^T : per-WG 128x128 tile, K = 128, async tile fill.
// ---------------------------------------------------------------------------
__global__ __launch_bounds__(256) void sim_kernel(
    const _Float16* __restrict__ Pw, const _Float16* __restrict__ Zn,
    float* __restrict__ out)
{
    __shared__ _Float16 Pa[128][136];   // rows x k (272B stride)
    __shared__ _Float16 Zt[128][136];   // cols x k

    const int ct = blockIdx.x, rt = blockIdx.y, b = blockIdx.z;
    const int row0 = rt * 128, col0 = ct * 128;
    const int tid = threadIdx.x, lane = tid & 31, w = tid >> 5;

    {
        int rr = tid >> 1, hb = (tid & 1) * 64;
        const _Float16* gp = Pw + ((size_t)b * NNODES + row0 + rr) * 128 + hb;
        const _Float16* gz = Zn + ((size_t)b * NNODES + col0 + rr) * 128 + hb;
        #pragma unroll
        for (int u = 0; u < 8; ++u) {
            async_copy16(gp + u * 8, &Pa[rr][hb + u * 8]);
            async_copy16(gz + u * 8, &Zt[rr][hb + u * 8]);
        }
    }
    async_wait();
    __syncthreads();

    v8f acc[8] = {v8f{}, v8f{}, v8f{}, v8f{}, v8f{}, v8f{}, v8f{}, v8f{}};
    const int lr = lane & 15;
    const int koff = (lane < 16) ? 0 : 8;
    const int r = (w << 4) + lr;

    #pragma unroll
    for (int kc = 0; kc < 128; kc += 32) {
        v16h a = load_frag(&Pa[r][kc], koff);
        #pragma unroll
        for (int t = 0; t < 8; ++t) {
            v16h bf = load_frag(&Zt[t * 16 + lr][kc], koff);
            acc[t] = wmma_f16(a, bf, acc[t]);
        }
    }

    const int rbase = row0 + (w << 4) + ((lane < 16) ? 0 : 8);
    #pragma unroll
    for (int t = 0; t < 8; ++t) {
        int col = col0 + t * 16 + lr;
        #pragma unroll
        for (int j = 0; j < 8; ++j)
            out[((size_t)b * NNODES + rbase + j) * NNODES + col] = acc[t][j];
    }
}

__global__ __launch_bounds__(256) void cvt_kernel(
    const float* __restrict__ in, _Float16* __restrict__ out, int n)
{
    int i = blockIdx.x * 256 + threadIdx.x;
    if (i < n) out[i] = (_Float16)in[i];
}

// ---------------------------------------------------------------------------
extern "C" void kernel_launch(void* const* d_in, const int* in_sizes, int n_in,
                              void* d_out, int out_size, void* d_ws, size_t ws_size,
                              hipStream_t stream) {
    const float* h0  = (const float*)d_in[0];   // [256,512,64]
    const float* A   = (const float*)d_in[1];   // [512,512]
    const float* np_ = (const float*)d_in[2];   // [2*64*64]
    const float* wv  = (const float*)d_in[3];   // [128]
    const float* t0p = (const float*)d_in[4];
    const float* t1p = (const float*)d_in[5];

    char* p = (char*)d_ws;
    float*    x_cur = (float*)p;     p += NELEM * 4;           // 32 MB
    float*    accb  = (float*)p;     p += NELEM * 4;           // 32 MB
    float*    Kbuf  = (float*)p;     p += NELEM * 4;           // 32 MB
    _Float16* Xh    = (_Float16*)p;  p += NELEM * 2;           // 16 MB
    _Float16* T1h   = (_Float16*)p;  p += NELEM * 2;           // 16 MB
    _Float16* Ah    = (_Float16*)p;  p += (size_t)NNODES * NNODES * 2;
    _Float16* Wh    = (_Float16*)p;  p += 2 * 64 * 64 * 2;
    // after the RK loop, acc/K buffers are free -> alias for similarity stage
    _Float16* Zn = (_Float16*)accb;  // B*N*128 halves == 32 MB
    _Float16* Pw = (_Float16*)Kbuf;

    (void)hipMemcpyAsync(x_cur, h0, NELEM * sizeof(float),
                         hipMemcpyDeviceToDevice, stream);
    cvt_kernel<<<(NNODES * NNODES) / 256, 256, 0, stream>>>(A, Ah, NNODES * NNODES);
    cvt_kernel<<<(2 * 64 * 64) / 256, 256, 0, stream>>>(np_, Wh, 2 * 64 * 64);
    cvt_kernel<<<(int)(NELEM / 256), 256, 0, stream>>>(h0, Xh, (int)NELEM);

    const float ca[4] = {1.f / 6.f, 1.f / 3.f, 1.f / 3.f, 1.f / 6.f};
    const float cb[4] = {0.5f, 0.5f, 1.0f, 0.0f};
    const int   md[4] = {0, 1, 1, 2};

    dim3 lgrid(2, BATCH);
    const int ublocks = (int)(NELEM / 4 / 256);

    for (int step = 0; step < 10; ++step) {
        for (int s = 0; s < 4; ++s) {
            layer_kernel<<<lgrid, 256, 0, stream>>>(Xh,  Ah, Wh,        nullptr, T1h);
            layer_kernel<<<lgrid, 256, 0, stream>>>(T1h, Ah, Wh + 4096, Kbuf,    nullptr);
            update_kernel<<<ublocks, 256, 0, stream>>>(x_cur, accb, Kbuf, Xh,
                                                       t0p, t1p, ca[s], cb[s], md[s]);
        }
    }

    norm_kernel<<<(BATCH * NNODES) / 8, 256, 0, stream>>>(h0, x_cur, wv, Zn, Pw);
    sim_kernel<<<dim3(4, 4, BATCH), 256, 0, stream>>>(Pw, Zn, (float*)d_out);
}
